// cls_model_18760417149417
// MI455X (gfx1250) — compile-verified, matrix operand order checked by reading, not run on hardware
//
#include <hip/hip_runtime.h>

#define B_ 4
#define N_ 4096
#define KNN 10
#define SLOPE 0.2f
#define BN_SCL 0.99999500003749969f   /* 1/sqrt(1+1e-5) */

typedef __attribute__((ext_vector_type(16))) _Float16 v16h;
typedef __attribute__((ext_vector_type(8)))  float    v8f;
typedef __attribute__((ext_vector_type(4))) unsigned int v4u;
typedef __attribute__((ext_vector_type(8)))  int      v8i;
typedef __attribute__((ext_vector_type(4)))  int      v4i;

// ---- WMMA fragment helpers -------------------------------------------------
// 16x32 (MxK) f16 A/B fragment from an LDS row. ISA layout: lanes 0-15 hold
// K 0..7 (v0..3) and 16..23 (v4..7); lanes 16-31 hold K 8..15 and 24..31.
static __device__ __forceinline__ v16h load_frag(const _Float16* rowp, int k0, int hi) {
  v16h f;
#pragma unroll
  for (int e = 0; e < 16; ++e) {
    int k = k0 + (e & 7) + hi + ((e & 8) ? 16 : 0);
    f[e] = rowp[k];
  }
  return f;
}

static __device__ __forceinline__ v8f wmma_f16(v16h a, v16h b, v8f c) {
  return __builtin_amdgcn_wmma_f32_16x16x32_f16(false, a, false, b, (short)0, c,
                                                false, false);
}

// ---- Tensor Data Mover: contiguous f16 global -> LDS copy ------------------
// D# per CDNA5 ISA ch.8: group0 = {count=1 | lds_addr | global_addr | type=2},
// group1 = {data_size=2B, tensor_dim0=tile_dim0=nelem, tensor_dim1=tile_dim1=1,
// dim0_stride=nelem}; groups 2/3 zero (<=2D tile). Tracked by TENSORcnt.
// This toolchain exposes the 6-arg builtin (extra int32x8 group before cpol).
static __device__ __forceinline__ void tdm_load_f16(unsigned lds_addr,
                                                    const _Float16* gsrc,
                                                    unsigned nelem) {
  unsigned long long ga = (unsigned long long)(const void*)gsrc;
  v4u g0;
  g0[0] = 1u;                                   // descriptor count = 1
  g0[1] = lds_addr;                             // lds_addr [63:32]
  g0[2] = (unsigned)ga;                         // global_addr [95:64]
  g0[3] = (unsigned)(ga >> 32) | (2u << 30);    // addr[56:32] | type=2 [127:126]
  v8i g1;
  g1[0] = 1 << 16;                              // data_size=1 (2 bytes), no mcast
  g1[1] = (int)((nelem & 0xFFFFu) << 16);       // tensor_dim0[15:0]  -> [63:48]
  g1[2] = (int)((nelem >> 16) & 0xFFFFu)        // tensor_dim0[31:16] -> [79:64]
        | (1 << 16);                            // tensor_dim1 = 1    -> [95:80]
  g1[3] = (int)((nelem & 0xFFFFu) << 16);       // tile_dim0 = nelem  -> [127:112]
  g1[4] = 1;                                    // tile_dim1 = 1      -> [143:128]
  g1[5] = (int)nelem;                           // tensor_dim0_stride -> [191:160]
  g1[6] = 0;
  g1[7] = 0;
  v4i z4 = {0, 0, 0, 0};
  v8i z8 = {0, 0, 0, 0, 0, 0, 0, 0};
  __builtin_amdgcn_tensor_load_to_lds(g0, g1, z4, z4, z8, 0);
}

static __device__ __forceinline__ unsigned lds_off(const void* p) {
  return (unsigned)(unsigned long long)(p);     // addrspace(3): low 32b = LDS offset
}

// ---------------------------------------------------------------- cvt + sqnorm
__global__ void cvt_kernel(const float* __restrict__ x, int ldx, int C, int cshift,
                           _Float16* __restrict__ xh, float* __restrict__ sq) {
  int i = blockIdx.x * 256 + threadIdx.x;
  if (i >= B_ * N_) return;
  const float* xr = x + (size_t)i * ldx;
  _Float16* xo = xh + ((size_t)i << cshift);
  int Cpad = 1 << cshift;
  float s = 0.f;
  for (int c = 0; c < Cpad; ++c) {
    float v = (c < C) ? xr[c] : 0.f;
    s += v * v;
    xo[c] = (_Float16)v;
  }
  sq[i] = s;
}

// ---------------------------------------------------------------- kNN (TDM + WMMA gram + top-k)
__global__ void knn_kernel(const _Float16* __restrict__ xh,
                           const float* __restrict__ sq,
                           int* __restrict__ nidx, int cshift) {
  __shared__ _Float16 arow[16 * 128];
  __shared__ _Float16 bcol[2][16 * 128];
  __shared__ float dtile[16][17];
  __shared__ float sqc[16];
  const int lane = threadIdx.x;
  const int b = blockIdx.y;
  const int m0 = blockIdx.x << 4;
  const int Cpad = 1 << cshift;
  const unsigned tile_h = 16u << cshift;        // f16 elems per 16-row tile
  const _Float16* xb = xh + (((size_t)b * N_) << cshift);

  // Async-stage A rows and the first B tile via the Tensor Data Mover.
  tdm_load_f16(lds_off(arow), xb + (((size_t)m0) << cshift), tile_h);
  tdm_load_f16(lds_off(&bcol[0][0]), xb, tile_h);

  float sqm = (lane < 16) ? sq[(size_t)b * N_ + m0 + lane] : 0.f;

  const int hi = (lane & 16) ? 8 : 0;
  const int nn = lane & 15;
  const int mb = (lane & 16) ? 8 : 0;
  const _Float16* arp = &arow[nn << cshift];

  float vals[KNN + 1];
  int   inds[KNN + 1];
#pragma unroll
  for (int s = 0; s <= KNN; ++s) { vals[s] = -3.0e38f; inds[s] = 0; }

  const int ntiles = N_ / 16;
  for (int t = 0; t < ntiles; ++t) {
    const int j0 = t << 4;
    const int cur = t & 1;
    if (t + 1 < ntiles)   // prefetch next B tile while computing this one
      tdm_load_f16(lds_off(&bcol[cur ^ 1][0]),
                   xb + (((size_t)(j0 + 16)) << cshift), tile_h);
    if (lane < 16) sqc[lane] = sq[(size_t)b * N_ + j0 + lane];
    // TDM ops complete in-order per wave: <=1 outstanding means current
    // buffer (and arow on the first trip) is resident.
    if (t + 1 < ntiles) __builtin_amdgcn_s_wait_tensorcnt(1);
    else                __builtin_amdgcn_s_wait_tensorcnt(0);
    v8f acc = {};
    const _Float16* brp = &bcol[cur][nn << cshift];
    for (int ks = 0; ks < Cpad; ks += 32)
      acc = wmma_f16(load_frag(arp, ks, hi), load_frag(brp, ks, hi), acc);
#pragma unroll
    for (int v = 0; v < 8; ++v) dtile[mb + v][nn] = acc[v];
    __syncthreads();            // dtile + sqc visible to the scan lanes
    if (lane < 16) {
      const int row = m0 + lane;
      for (int jj = 0; jj < 16; ++jj) {
        int j = j0 + jj;
        float d = 2.f * dtile[lane][jj] - sqm - sqc[jj];
        if (j == row) d = 3.0e38f;             // self is always rank-0 (dropped)
        if (d > vals[KNN]) {
          vals[KNN] = d; inds[KNN] = j;
#pragma unroll
          for (int s = KNN; s > 0; --s) {      // bubble the new entry up (stable)
            if (vals[s] > vals[s - 1]) {
              float tv = vals[s]; vals[s] = vals[s - 1]; vals[s - 1] = tv;
              int   ti = inds[s]; inds[s] = inds[s - 1]; inds[s - 1] = ti;
            }
          }
        }
      }
    }
  }
  if (lane < 16) {
#pragma unroll
    for (int t = 0; t < KNN; ++t)
      nidx[((size_t)b * N_ + m0 + lane) * KNN + t] = inds[t + 1];
  }
}

// ---------------------------------------------------------------- EdgeConv (gather + WMMA + bn/lrelu/max)
__global__ void edge_conv_kernel(const float* __restrict__ x, int ldx, int C, int k2shift,
                                 const int* __restrict__ nidx,
                                 const float* __restrict__ w,
                                 const float* __restrict__ gamma,
                                 const float* __restrict__ beta,
                                 float* __restrict__ out,
                                 _Float16* __restrict__ outh,   // f16 mirror for TDM consumers
                                 int ldo, int O) {
  extern __shared__ char smem[];
  const int K2 = 1 << k2shift;
  _Float16* E    = (_Float16*)smem;              // 160 x K2 edge features (f16)
  _Float16* wt   = E + 160 * K2;                 // 16 x K2 weight tile (f16)
  float*    hbuf = (float*)(wt + 16 * K2);       // 160 x 16 raw GEMM results
  const int lane = threadIdx.x;
  const int b = blockIdx.y;
  const int p0 = blockIdx.x << 4;

  // Build edge matrix: row r = p*10+j -> [neigh - ctr | ctr], zero-padded to K2
  for (int r = lane; r < 16 * KNN; r += 32) {
    int p = r / KNN, j = r - p * KNN;
    int prow = p0 + p;
    int nb = nidx[((size_t)b * N_ + prow) * KNN + j];
    const float* xc = x + ((size_t)b * N_ + prow) * ldx;
    const float* xn = x + ((size_t)b * N_ + nb) * ldx;
    _Float16* er = E + r * K2;
    for (int c = 0; c < C; ++c) {
      float cv = xc[c], nv = xn[c];
      er[c]     = (_Float16)(nv - cv);
      er[C + c] = (_Float16)cv;
    }
    for (int c = 2 * C; c < K2; ++c) er[c] = (_Float16)0.f;
  }

  const int hi = (lane & 16) ? 8 : 0;
  const int nn = lane & 15;
  const int mb = (lane & 16) ? 8 : 0;
  const _Float16* brp = wt + (nn << k2shift);

  for (int o0 = 0; o0 < O; o0 += 16) {
    __syncthreads();
    for (int t = lane; t < (16 << k2shift); t += 32) {
      int n = t >> k2shift, k = t & (K2 - 1);
      wt[t] = (k < 2 * C) ? (_Float16)w[(size_t)(o0 + n) * (2 * C) + k]
                          : (_Float16)0.f;
    }
    __syncthreads();
    for (int mt = 0; mt < KNN; ++mt) {           // 160 rows = 10 M-tiles of 16
      v8f acc = {};
      const _Float16* arp = E + ((mt * 16 + nn) << k2shift);
      for (int ks = 0; ks < K2; ks += 32)
        acc = wmma_f16(load_frag(arp, ks, hi), load_frag(brp, ks, hi), acc);
#pragma unroll
      for (int v = 0; v < 8; ++v) hbuf[(mt * 16 + mb + v) * 16 + nn] = acc[v];
    }
    __syncthreads();
    // bn + leaky-relu + max over the 10 neighbors of each point
    for (int t = lane; t < 256; t += 32) {
      int p = t >> 4, ch = t & 15;
      float g  = gamma[o0 + ch] * BN_SCL;
      float be = beta[o0 + ch];
      float mx = -3.0e38f;
      for (int j = 0; j < KNN; ++j) {
        float h = hbuf[(p * KNN + j) * 16 + ch] * g + be;
        h = (h > 0.f) ? h : SLOPE * h;
        mx = fmaxf(mx, h);
      }
      size_t oidx = ((size_t)b * N_ + p0 + p) * ldo + o0 + ch;
      out[oidx]  = mx;
      outh[oidx] = (_Float16)mx;
    }
  }
}

// ---------------------------------------------------------------- cat @ w5^T + bn/lrelu + max over N
__global__ void gemm5_max_kernel(const _Float16* __restrict__ cath,
                                 const float* __restrict__ w5,
                                 const float* __restrict__ g5,
                                 const float* __restrict__ b5,
                                 float* __restrict__ pool) {
  __shared__ _Float16 wt[16 * 512];
  __shared__ _Float16 at[2][16 * 512];
  const int lane = threadIdx.x;
  const int b = blockIdx.y;
  const int o0 = blockIdx.x << 4;
  for (int t = lane; t < 16 * 512; t += 32) {
    int n = t >> 9, k = t & 511;
    wt[t] = (_Float16)w5[(size_t)(o0 + n) * 512 + k];
  }
  const _Float16* cb = cath + (size_t)b * N_ * 512;
  tdm_load_f16(lds_off(&at[0][0]), cb, 16 * 512);   // first A tile via TDM
  __syncthreads();                                   // wt visible

  const int hi = (lane & 16) ? 8 : 0;
  const int nn = lane & 15;
  const float g  = g5[o0 + nn] * BN_SCL;
  const float be = b5[o0 + nn];
  const _Float16* brp = &wt[nn * 512];
  float rmax = -3.0e38f;
  const int ntiles = N_ / 16;
  for (int t = 0; t < ntiles; ++t) {
    const int cur = t & 1;
    if (t + 1 < ntiles)
      tdm_load_f16(lds_off(&at[cur ^ 1][0]), cb + (size_t)(t + 1) * 16 * 512,
                   16 * 512);
    if (t + 1 < ntiles) __builtin_amdgcn_s_wait_tensorcnt(1);
    else                __builtin_amdgcn_s_wait_tensorcnt(0);
    v8f acc = {};
    const _Float16* arp = &at[cur][nn * 512];
    for (int ks = 0; ks < 512; ks += 32)
      acc = wmma_f16(load_frag(arp, ks, hi), load_frag(brp, ks, hi), acc);
#pragma unroll
    for (int v = 0; v < 8; ++v) {
      float h = acc[v] * g + be;
      h = (h > 0.f) ? h : SLOPE * h;
      rmax = fmaxf(rmax, h);
    }
  }
  rmax = fmaxf(rmax, __shfl_xor(rmax, 16, 32));  // merge lane halves (same channel)
  if (lane < 16) pool[(size_t)b * 1024 + o0 + lane] = rmax;
}

// ---------------------------------------------------------------- MLP head (tiny, VALU)
__global__ void head_kernel(const float* __restrict__ pool,
                            const float* __restrict__ fw1, const float* __restrict__ fg1,
                            const float* __restrict__ fbt1,
                            const float* __restrict__ fw2, const float* __restrict__ fb2,
                            const float* __restrict__ fg2, const float* __restrict__ fbt2,
                            const float* __restrict__ fw3, const float* __restrict__ fb3,
                            float* __restrict__ out) {
  __shared__ float h0[B_ * 1024];
  __shared__ float h1[B_ * 512];
  __shared__ float h2[B_ * 256];
  const int t = threadIdx.x;   // 256
  for (int i = t; i < B_ * 1024; i += 256) h0[i] = pool[i];
  __syncthreads();
  for (int i = t; i < B_ * 512; i += 256) {
    int bb = i >> 9, o = i & 511;
    float acc = 0.f;
    for (int k = 0; k < 1024; ++k) acc += h0[bb * 1024 + k] * fw1[(size_t)o * 1024 + k];
    acc = acc * (fg1[o] * BN_SCL) + fbt1[o];
    h1[i] = (acc > 0.f) ? acc : SLOPE * acc;
  }
  __syncthreads();
  for (int i = t; i < B_ * 256; i += 256) {
    int bb = i >> 8, o = i & 255;
    float acc = fb2[o];
    for (int k = 0; k < 512; ++k) acc += h1[bb * 512 + k] * fw2[(size_t)o * 512 + k];
    acc = acc * (fg2[o] * BN_SCL) + fbt2[o];
    h2[i] = (acc > 0.f) ? acc : SLOPE * acc;
  }
  __syncthreads();
  if (t < B_ * 3) {
    int bb = t / 3, o = t - bb * 3;
    float acc = fb3[o];
    for (int k = 0; k < 256; ++k) acc += h2[bb * 256 + k] * fw3[(size_t)o * 256 + k];
    out[t] = acc;
  }
}

// ---------------------------------------------------------------- driver
extern "C" void kernel_launch(void* const* d_in, const int* in_sizes, int n_in,
                              void* d_out, int out_size, void* d_ws, size_t ws_size,
                              hipStream_t stream) {
  (void)in_sizes; (void)n_in; (void)out_size; (void)ws_size;
  const float* points = (const float*)d_in[0];
  // d_in[1] = k (always 10)
  const float* w1 = (const float*)d_in[2];
  const float* g1 = (const float*)d_in[3];
  const float* b1 = (const float*)d_in[4];
  const float* w2 = (const float*)d_in[5];
  const float* g2 = (const float*)d_in[6];
  const float* b2 = (const float*)d_in[7];
  const float* w3 = (const float*)d_in[8];
  const float* g3 = (const float*)d_in[9];
  const float* b3 = (const float*)d_in[10];
  const float* w4 = (const float*)d_in[11];
  const float* g4 = (const float*)d_in[12];
  const float* b4 = (const float*)d_in[13];
  const float* w5 = (const float*)d_in[14];
  const float* g5 = (const float*)d_in[15];
  const float* b5 = (const float*)d_in[16];
  const float* fw1  = (const float*)d_in[17];
  const float* fg1  = (const float*)d_in[18];
  const float* fbt1 = (const float*)d_in[19];
  const float* fw2  = (const float*)d_in[20];
  const float* fb2  = (const float*)d_in[21];
  const float* fg2  = (const float*)d_in[22];
  const float* fbt2 = (const float*)d_in[23];
  const float* fw3  = (const float*)d_in[24];
  const float* fb3  = (const float*)d_in[25];

  char* ws = (char*)d_ws;
  float* cat = (float*)ws;            ws += (size_t)B_ * N_ * 512 * sizeof(float);
  _Float16* cath = (_Float16*)ws;     ws += (size_t)B_ * N_ * 512 * sizeof(_Float16);
  _Float16* xh = (_Float16*)ws;       ws += (size_t)B_ * N_ * 128 * sizeof(_Float16);
  float* sq = (float*)ws;             ws += (size_t)B_ * N_ * sizeof(float);
  int* nidx = (int*)ws;               ws += (size_t)B_ * N_ * KNN * sizeof(int);
  float* pool = (float*)ws;           ws += (size_t)B_ * 1024 * sizeof(float);

  struct Layer {
    const float* x; int ldx, C, cshift, k2shift, O, ooff;
    const float *w, *g, *bt;
  };
  const Layer L[4] = {
    { points,    3,   3, 5, 5,  64,   0, w1, g1, b1 },
    { cat + 0,  512,  64, 6, 7,  64,  64, w2, g2, b2 },
    { cat + 64, 512,  64, 6, 7, 128, 128, w3, g3, b3 },
    { cat + 128,512, 128, 7, 8, 256, 256, w4, g4, b4 },
  };

  const dim3 tile_grid(N_ / 16, B_);
  for (int i = 0; i < 4; ++i) {
    cvt_kernel<<<dim3((B_ * N_ + 255) / 256), 256, 0, stream>>>(
        L[i].x, L[i].ldx, L[i].C, L[i].cshift, xh, sq);
    knn_kernel<<<tile_grid, 32, 0, stream>>>(xh, sq, nidx, L[i].cshift);
    size_t smem = (size_t)(160 + 16) * ((size_t)1 << L[i].k2shift) * sizeof(_Float16)
                + (size_t)160 * 16 * sizeof(float);
    edge_conv_kernel<<<tile_grid, 32, smem, stream>>>(
        L[i].x, L[i].ldx, L[i].C, L[i].k2shift, nidx,
        L[i].w, L[i].g, L[i].bt, cat + L[i].ooff, cath + L[i].ooff, 512, L[i].O);
  }

  gemm5_max_kernel<<<dim3(1024 / 16, B_), 32, 0, stream>>>(cath, w5, g5, b5, pool);
  head_kernel<<<1, 256, 0, stream>>>(pool, fw1, fg1, fbt1, fw2, fb2, fg2, fbt2,
                                     fw3, fb3, (float*)d_out);
}
